// AdvancedCardiomyocyteGNN_34763465293824
// MI455X (gfx1250) — matrix operand order repaired and checked by reading, not compiled, not run on hardware
//
#include <hip/hip_runtime.h>
#include <hip/hip_fp16.h>

// ---------------------------------------------------------------------------
// AdvancedCardiomyocyteGNN forward on MI455X (gfx1250, wave32, WMMA).
//  - Dense GEMMs: v_wmma_f32_16x16x32_f16, weights pre-transposed to f16 so
//    both A and B fragments are contiguous 32B loads.
//  - B panel staged to LDS with global_load_async_to_lds_b128 (double
//    buffered, s_wait_asynccnt) when the toolchain exposes the builtins.
//  - Edge scatters: float4-vectorized global f32 atomics (L2-resident).
//  - LayerNorm: one wave32 per node, shfl_xor tree reduction.
// ---------------------------------------------------------------------------

typedef __attribute__((ext_vector_type(16))) _Float16 v16h;
typedef __attribute__((ext_vector_type(8)))  _Float16 v8h;
typedef __attribute__((ext_vector_type(8)))  float    v8f;
typedef __attribute__((ext_vector_type(4)))  int      v4i;

#if defined(__gfx1250__) && __has_builtin(__builtin_amdgcn_global_load_async_to_lds_b128) && __has_builtin(__builtin_amdgcn_s_wait_asynccnt)
#define CDNA5_ASYNC 1
typedef __attribute__((address_space(1))) v4i gv4i;   // global int4
typedef __attribute__((address_space(3))) v4i lv4i;   // LDS int4
#else
#define CDNA5_ASYNC 0
#endif

// ------------------------------- utility -----------------------------------

__global__ void fill_f32_kernel(float* p, float v, long long n) {
    long long i = (long long)blockIdx.x * blockDim.x + threadIdx.x;
    if (i < n) p[i] = v;
}

// convert f32 W[k,m] -> f16 Bt[m,k] (transposed, so GEMM B reads contiguous)
__global__ void f16_transpose_kernel(const float* W, _Float16* Bt, int k, int m) {
    int idx = blockIdx.x * blockDim.x + threadIdx.x;
    if (idx >= k * m) return;
    int r = idx / m, c = idx % m;
    Bt[(size_t)c * k + r] = (_Float16)W[idx];
}

// BN (eval) over feature dim, output f16
__global__ void bn_f16_kernel(const float* in, const float* g, const float* b,
                              const float* rm, const float* rv, _Float16* out,
                              long long total, int C) {
    long long i = (long long)blockIdx.x * blockDim.x + threadIdx.x;
    if (i >= total) return;
    int c = (int)(i % C);
    float v = (in[i] - rm[c]) * rsqrtf(rv[c] + 1e-5f) * g[c] + b[c];
    out[i] = (_Float16)v;
}

// BN (eval) + relu, f32 out (classifier)
__global__ void bn_relu_f32_kernel(const float* in, const float* g, const float* b,
                                   const float* rm, const float* rv, float* out,
                                   long long total, int C) {
    long long i = (long long)blockIdx.x * blockDim.x + threadIdx.x;
    if (i >= total) return;
    int c = (int)(i % C);
    float v = (in[i] - rm[c]) * rsqrtf(rv[c] + 1e-5f) * g[c] + b[c];
    out[i] = fmaxf(v, 0.f);
}

// ------------------------------- WMMA GEMM ---------------------------------
// C[n,m] = act(A[n,k] @ B[k,m] + bias), with B supplied TRANSPOSED as
// Bt[m,k] (f16). Block = 256 threads = 8 waves; wave w owns row tile
// (blockIdx.x*8+w)*16; block owns NCT col tiles of 16 at blockIdx.y*16*NCT.
// n % 16 == 0, k % 32 == 0, m % (16*NCT) == 0 assumed.
template <int NCT>
__global__ __launch_bounds__(256) void wmma_gemm_kernel(
    const _Float16* __restrict__ A, const _Float16* __restrict__ Bt,
    const float* __restrict__ bias, float* __restrict__ C,
    _Float16* __restrict__ Ch, int n, int k, int m, int relu) {
    const int tid = threadIdx.x;
    const int wave = tid >> 5, lane = tid & 31;
    const int lm = lane & 15, hi = lane >> 4;
    int row0 = (blockIdx.x * 8 + wave) * 16;
    const bool active = row0 < n;
    if (!active) row0 = n - 16;           // clamp: compute duplicate tile, mask store
    const int col0 = blockIdx.y * (16 * NCT);

    v8f z = {0.f,0.f,0.f,0.f,0.f,0.f,0.f,0.f};
    v8f acc[NCT];
#pragma unroll
    for (int t = 0; t < NCT; ++t) acc[t] = z;

    const _Float16* Arow = A + (size_t)(row0 + lm) * k;
    const int nsteps = k / 32;

#if CDNA5_ASYNC
    // ---- async double-buffered B panel staging: Bs[phase][col][32 K-halfs] ----
    __shared__ __align__(64) _Float16 Bs[2][16 * NCT * 32];
    auto stage = [&](int step, int phase) {
        if (tid < 64 * NCT) {             // wave-uniform split (multiples of 128)
            int c = tid >> 2, part = tid & 3;
            _Float16* gp = const_cast<_Float16*>(
                Bt + (size_t)(col0 + c) * k + step * 32 + part * 8);
            __builtin_amdgcn_global_load_async_to_lds_b128(
                (gv4i*)gp, (lv4i*)&Bs[phase][c * 32 + part * 8], 0, 0);
        }
    };
    stage(0, 0);
    for (int s = 0; s < nsteps; ++s) {
        const int phase = s & 1;
        if (s + 1 < nsteps) {
            stage(s + 1, phase ^ 1);
            __builtin_amdgcn_s_wait_asynccnt(1);   // our phase's batch complete
        } else {
            __builtin_amdgcn_s_wait_asynccnt(0);
        }
        __syncthreads();
        const int kk = s * 32;
        if (kk + 32 < k) __builtin_prefetch(Arow + kk + 32, 0, 1); // global_prefetch_b8
        // A fragment (16x32): lanes0-15 K=0..15, lanes16-31 K=16..31 (contiguous)
        v16h a = *(const v16h*)(Arow + kk + hi * 16);
#pragma unroll
        for (int t = 0; t < NCT; ++t) {
            // B fragment from LDS: col = t*16+lm, halfs K = hi*16 + j (contiguous)
            v16h b = *(const v16h*)(&Bs[phase][(t * 16 + lm) * 32 + hi * 16]);
            acc[t] = __builtin_amdgcn_wmma_f32_16x16x32_f16(
                false, a, false, b, (short)0, acc[t], false, false);
        }
        __syncthreads();                  // buffer[phase] free for step s+2
    }
#else
    // ---- fallback: direct contiguous loads from transposed B ----
    for (int s = 0; s < nsteps; ++s) {
        const int kk = s * 32;
        if (kk + 32 < k) __builtin_prefetch(Arow + kk + 32, 0, 1);
        v16h a = *(const v16h*)(Arow + kk + hi * 16);
#pragma unroll
        for (int t = 0; t < NCT; ++t) {
            const _Float16* Brow = Bt + (size_t)(col0 + t * 16 + lm) * k + kk + hi * 16;
            v16h b = *(const v16h*)(Brow);
            acc[t] = __builtin_amdgcn_wmma_f32_16x16x32_f16(
                false, a, false, b, (short)0, acc[t], false, false);
        }
    }
#endif

    if (active) {
#pragma unroll
        for (int t = 0; t < NCT; ++t) {
            int c = col0 + t * 16 + lm;
            float bv = bias ? bias[c] : 0.f;
#pragma unroll
            for (int i = 0; i < 8; ++i) {
                int r = row0 + hi * 8 + i;    // C/D: M = i + 8*hi, N = lane%16
                float v = acc[t][i] + bv;
                if (relu) v = fmaxf(v, 0.f);
                C[(size_t)r * m + c] = v;
                if (Ch) Ch[(size_t)r * m + c] = (_Float16)v;
            }
        }
    }
}

// ------------------------------- GAT pieces --------------------------------

__device__ inline void atomicMaxF(float* addr, float val) {
    unsigned int* ua = (unsigned int*)addr;
    unsigned int old = __float_as_uint(*addr);
    while (__uint_as_float(old) < val) {
        unsigned int assumed = old;
        old = atomicCAS(ua, assumed, __float_as_uint(val));
        if (old == assumed) break;
    }
}

__global__ void gat_scores_kernel(const float* __restrict__ h,
                                  const float* __restrict__ asv,
                                  const float* __restrict__ adv,
                                  float* ssrc, float* sdst, int n, int H, int C) {
    int idx = blockIdx.x * blockDim.x + threadIdx.x;
    if (idx >= n * H) return;
    int hh = idx % H;
    const float* hp = h + (size_t)idx * C;
    const float* ap = asv + hh * C;
    const float* dp = adv + hh * C;
    float s1 = 0.f, s2 = 0.f;
    for (int c = 0; c < C; ++c) { float v = hp[c]; s1 += v * ap[c]; s2 += v * dp[c]; }
    ssrc[idx] = s1; sdst[idx] = s2;
}

__device__ inline void edge_decode(const int* ei, int E, int e, int& s, int& d) {
    if (e < E) { s = ei[e]; d = ei[E + e]; } else { s = d = e - E; }
}

__global__ void gat_edge_max_kernel(const int* ei, int E, int ET, int H,
                                    const float* ssrc, const float* sdst, float* mx) {
    int idx = blockIdx.x * blockDim.x + threadIdx.x;
    if (idx >= ET * H) return;
    int e = idx / H, hh = idx % H, s, d;
    edge_decode(ei, E, e, s, d);
    float v = ssrc[s * H + hh] + sdst[d * H + hh];
    v = v > 0.f ? v : 0.2f * v;
    atomicMaxF(&mx[d * H + hh], v);
}

__global__ void gat_edge_exp_kernel(const int* ei, int E, int ET, int H,
                                    const float* ssrc, const float* sdst,
                                    const float* mx, float* alpha, float* ssum) {
    int idx = blockIdx.x * blockDim.x + threadIdx.x;
    if (idx >= ET * H) return;
    int e = idx / H, hh = idx % H, s, d;
    edge_decode(ei, E, e, s, d);
    float v = ssrc[s * H + hh] + sdst[d * H + hh];
    v = v > 0.f ? v : 0.2f * v;
    float ex = expf(v - mx[d * H + hh]);
    alpha[idx] = ex;
    atomicAdd(&ssum[d * H + hh], ex);
}

__global__ void gat_alpha_norm_kernel(const int* ei, int E, int ET, int H,
                                      const float* ssum, float* alpha) {
    int idx = blockIdx.x * blockDim.x + threadIdx.x;
    if (idx >= ET * H) return;
    int e = idx / H, hh = idx % H, s, d;
    edge_decode(ei, E, e, s, d);
    alpha[idx] /= (ssum[d * H + hh] + 1e-16f);
}

// 4 channels per thread: one edge decode + alpha load per quad
__global__ void gat_agg_kernel(const int* ei, int E, int ET, int H, int C,
                               const float* __restrict__ h,
                               const float* __restrict__ alpha, float* agg) {
    long long idx = (long long)blockIdx.x * blockDim.x + threadIdx.x;
    const int HC = H * C, Q = HC >> 2;
    long long tot = (long long)ET * Q;
    if (idx >= tot) return;
    int e = (int)(idx / Q);
    int q = (int)(idx % Q);
    int hc = q << 2, hh = hc / C, s, d;
    edge_decode(ei, E, e, s, d);
    float al = alpha[e * H + hh];
    const float4 hv = *(const float4*)(h + (size_t)s * HC + hc);
    float* ap = agg + (size_t)d * HC + hc;
    atomicAdd(ap + 0, hv.x * al);
    atomicAdd(ap + 1, hv.y * al);
    atomicAdd(ap + 2, hv.z * al);
    atomicAdd(ap + 3, hv.w * al);
}

__global__ void gat_post_kernel(const float* agg, const float* bias, float* out,
                                int n, int H, int C) {
    int idx = blockIdx.x * blockDim.x + threadIdx.x;
    if (idx >= n * C) return;
    int node = idx / C, c = idx % C;
    float s = 0.f;
    for (int hh = 0; hh < H; ++hh) s += agg[((size_t)node * H + hh) * C + c];
    out[idx] = s / H + bias[c];
}

// ------------------------------- GCN pieces --------------------------------

__global__ void gcn_deg_kernel(const int* ei, int E, int ET, float* deg) {
    int e = blockIdx.x * blockDim.x + threadIdx.x;
    if (e >= ET) return;
    int s, d; edge_decode(ei, E, e, s, d);
    atomicAdd(&deg[d], 1.0f);
}

__global__ void gcn_dis_kernel(const float* deg, float* dis, int n) {
    int i = blockIdx.x * blockDim.x + threadIdx.x;
    if (i >= n) return;
    float dg = deg[i];
    dis[i] = dg > 0.f ? rsqrtf(fmaxf(dg, 1e-12f)) : 0.f;
}

__global__ void gcn_agg_kernel(const int* ei, int E, int ET, int C,
                               const float* __restrict__ h,
                               const float* __restrict__ dis, float* agg) {
    long long idx = (long long)blockIdx.x * blockDim.x + threadIdx.x;
    const int Q = C >> 2;
    long long tot = (long long)ET * Q;
    if (idx >= tot) return;
    int e = (int)(idx / Q), c = (int)(idx % Q) << 2, s, d;
    edge_decode(ei, E, e, s, d);
    float w = dis[s] * dis[d];
    const float4 hv = *(const float4*)(h + (size_t)s * C + c);
    float* ap = agg + (size_t)d * C + c;
    atomicAdd(ap + 0, hv.x * w);
    atomicAdd(ap + 1, hv.y * w);
    atomicAdd(ap + 2, hv.z * w);
    atomicAdd(ap + 3, hv.w * w);
}

__global__ void gcn_post_kernel(const float* agg, const float* bias,
                                _Float16* h16a, int lda, _Float16* h16b, int ldb,
                                int n, int C) {
    int idx = blockIdx.x * blockDim.x + threadIdx.x;
    if (idx >= n * C) return;
    int r = idx / C, c = idx % C;
    float v = fmaxf(agg[idx] + bias[c], 0.f);
    if (h16a) h16a[(size_t)r * lda + c] = (_Float16)v;
    if (h16b) h16b[(size_t)r * ldb + c] = (_Float16)v;
}

// ------------------------------- LayerNorm ---------------------------------

__global__ __launch_bounds__(256) void ln_kernel(
    const float* __restrict__ in, const float* __restrict__ add,
    const float* __restrict__ g, const float* __restrict__ b,
    float* out, _Float16* outh, int n, int C, int relu) {
    int wave = threadIdx.x >> 5, lane = threadIdx.x & 31;
    int node = blockIdx.x * 8 + wave;
    if (node >= n) return;
    const float* ip = in + (size_t)node * C;
    const float* ap = add ? add + (size_t)node * C : nullptr;
    int cnt = C / 32;              // C in {64,128}
    float vals[4];
    float s = 0.f, s2 = 0.f;
    for (int j = 0; j < cnt; ++j) {
        float v = ip[lane + j * 32];
        if (ap) v += ap[lane + j * 32];
        vals[j] = v; s += v; s2 += v * v;
    }
    for (int o = 16; o > 0; o >>= 1) {
        s  += __shfl_xor(s,  o, 32);
        s2 += __shfl_xor(s2, o, 32);
    }
    float mu = s / C;
    float var = s2 / C - mu * mu;
    float inv = rsqrtf(var + 1e-5f);
    for (int j = 0; j < cnt; ++j) {
        int c = lane + j * 32;
        float v = (vals[j] - mu) * inv * g[c] + b[c];
        if (relu) v = fmaxf(v, 0.f);
        if (out)  out [(size_t)node * C + c] = v;
        if (outh) outh[(size_t)node * C + c] = (_Float16)v;
    }
}

// ------------------------------- classifier tail ---------------------------

__global__ void final_gemm_kernel(const float* h, const float* W, const float* b,
                                  float* out, int N) {
    int idx = blockIdx.x * blockDim.x + threadIdx.x;
    if (idx >= N * 5) return;
    int n = idx / 5, j = idx % 5;
    float s = b[j];
    for (int i = 0; i < 32; ++i) s += h[n * 32 + i] * W[i * 5 + j];
    out[idx] = s;
}

// ---------------------------------------------------------------------------

extern "C" void kernel_launch(void* const* d_in, const int* in_sizes, int n_in,
                              void* d_out, int out_size, void* d_ws, size_t ws_size,
                              hipStream_t stream) {
    const float* x   = (const float*)d_in[0];
    const int*   ei  = (const int*)  d_in[1];
    const float* gin = (const float*)d_in[2];
    const float* bti = (const float*)d_in[3];
    const float* rmi = (const float*)d_in[4];
    const float* rvi = (const float*)d_in[5];
    const float* W1  = (const float*)d_in[6];
    const float* as1 = (const float*)d_in[7];
    const float* ad1 = (const float*)d_in[8];
    const float* b1  = (const float*)d_in[9];
    const float* Wg1 = (const float*)d_in[10];
    const float* bg1 = (const float*)d_in[11];
    const float* W2  = (const float*)d_in[12];
    const float* as2 = (const float*)d_in[13];
    const float* ad2 = (const float*)d_in[14];
    const float* b2  = (const float*)d_in[15];
    const float* Wg2 = (const float*)d_in[16];
    const float* bg2 = (const float*)d_in[17];
    const float* Wsk = (const float*)d_in[18];
    const float* bsk = (const float*)d_in[19];
    const float* Wf  = (const float*)d_in[20];
    const float* bf  = (const float*)d_in[21];
    const float* g1  = (const float*)d_in[22];
    const float* be1 = (const float*)d_in[23];
    const float* g2  = (const float*)d_in[24];
    const float* be2 = (const float*)d_in[25];
    const float* g3  = (const float*)d_in[26];
    const float* be3 = (const float*)d_in[27];
    const float* Wc1 = (const float*)d_in[28];
    const float* bc1 = (const float*)d_in[29];
    const float* gbn = (const float*)d_in[30];
    const float* bbn = (const float*)d_in[31];
    const float* rmb = (const float*)d_in[32];
    const float* rvb = (const float*)d_in[33];
    const float* Wc2 = (const float*)d_in[34];
    const float* bc2 = (const float*)d_in[35];

    const int F = 512, C1 = 128, C2 = 64;
    const int N  = in_sizes[0] / F;
    const int E  = in_sizes[1] / 2;
    const int ET = E + N;                          // edges + self loops

    // ---- workspace carve-out (256B aligned) ----
    size_t off = 0;
    char* base = (char*)d_ws;
    auto alloc = [&](size_t bytes) -> void* {
        off = (off + 255) & ~(size_t)255;
        void* p = base + off; off += bytes; return p;
    };
    _Float16* xn_h  = (_Float16*)alloc((size_t)N * F * 2);
    _Float16* W1t   = (_Float16*)alloc((size_t)F * 512 * 2);   // [m,k] transposed
    _Float16* Wg1t  = (_Float16*)alloc(128 * 128 * 2);
    _Float16* W2t   = (_Float16*)alloc(128 * 128 * 2);
    _Float16* Wg2t  = (_Float16*)alloc(64 * 64 * 2);
    _Float16* Wskt  = (_Float16*)alloc(512 * 64 * 2);
    _Float16* Wft   = (_Float16*)alloc(192 * 64 * 2);
    _Float16* Wc1t  = (_Float16*)alloc(64 * 32 * 2);
    float* bigA  = (float*)alloc((size_t)N * 512 * 4);   // gemm outputs (h buffers)
    float* bigB  = (float*)alloc((size_t)N * 512 * 4);   // scatter accumulators
    float* ssrc  = (float*)alloc((size_t)N * 4 * 4);
    float* sdst  = (float*)alloc((size_t)N * 4 * 4);
    float* mx    = (float*)alloc((size_t)N * 4 * 4);
    float* ssum  = (float*)alloc((size_t)N * 4 * 4);
    float* alpha = (float*)alloc((size_t)ET * 4 * 4);
    float* x1pre = (float*)alloc((size_t)N * C1 * 4);
    _Float16* x1_h  = (_Float16*)alloc((size_t)N * C1 * 2);
    float* deg   = (float*)alloc((size_t)N * 4);
    float* dis   = (float*)alloc((size_t)N * 4);
    _Float16* x1g_h = (_Float16*)alloc((size_t)N * C1 * 2);
    _Float16* cat_h = (_Float16*)alloc((size_t)N * 192 * 2);
    float* x2pre = (float*)alloc((size_t)N * C2 * 4);
    _Float16* x2_h  = (_Float16*)alloc((size_t)N * C2 * 2);
    float* skipb = (float*)alloc((size_t)N * C2 * 4);
    float* fusedb= (float*)alloc((size_t)N * C2 * 4);
    _Float16* fin_h = (_Float16*)alloc((size_t)N * C2 * 2);
    float* hc1   = (float*)alloc((size_t)N * 32 * 4);
    float* hc1r  = (float*)alloc((size_t)N * 32 * 4);

    auto cdiv = [](long long a, long long b) { return (int)((a + b - 1) / b); };

    auto gemm4 = [&](const _Float16* A, const _Float16* Bt, const float* bias,
                     float* C, _Float16* Ch, int n, int k, int m, int relu) {
        dim3 g(cdiv(n / 16, 8), m / 64);
        wmma_gemm_kernel<4><<<g, 256, 0, stream>>>(A, Bt, bias, C, Ch, n, k, m, relu);
    };
    auto gemm2 = [&](const _Float16* A, const _Float16* Bt, const float* bias,
                     float* C, _Float16* Ch, int n, int k, int m, int relu) {
        dim3 g(cdiv(n / 16, 8), m / 32);
        wmma_gemm_kernel<2><<<g, 256, 0, stream>>>(A, Bt, bias, C, Ch, n, k, m, relu);
    };

    // ---- weights -> f16 transposed ----
    f16_transpose_kernel<<<cdiv((long long)F * 512, 256), 256, 0, stream>>>(W1, W1t, F, 512);
    f16_transpose_kernel<<<cdiv(128 * 128, 256), 256, 0, stream>>>(Wg1, Wg1t, 128, 128);
    f16_transpose_kernel<<<cdiv(128 * 128, 256), 256, 0, stream>>>(W2, W2t, 128, 128);
    f16_transpose_kernel<<<cdiv(64 * 64, 256), 256, 0, stream>>>(Wg2, Wg2t, 64, 64);
    f16_transpose_kernel<<<cdiv(512 * 64, 256), 256, 0, stream>>>(Wsk, Wskt, 512, 64);
    f16_transpose_kernel<<<cdiv(192 * 64, 256), 256, 0, stream>>>(Wf, Wft, 192, 64);
    f16_transpose_kernel<<<cdiv(64 * 32, 256), 256, 0, stream>>>(Wc1, Wc1t, 64, 32);

    // ---- input BN -> f16 ----
    bn_f16_kernel<<<cdiv((long long)N * F, 256), 256, 0, stream>>>(
        x, gin, bti, rmi, rvi, xn_h, (long long)N * F, F);

    // ---- degree / normalization (graph constant; shared by both GCNs) ----
    (void)hipMemsetAsync(deg, 0, (size_t)N * 4, stream);
    gcn_deg_kernel<<<cdiv(ET, 256), 256, 0, stream>>>(ei, E, ET, deg);
    gcn_dis_kernel<<<cdiv(N, 256), 256, 0, stream>>>(deg, dis, N);

    auto gat_stage = [&](const float* hbuf, const float* asv, const float* adv,
                         const float* bias, int H, int C, float* aggbuf, float* outpre) {
        fill_f32_kernel<<<cdiv(N * H, 256), 256, 0, stream>>>(mx, -1e30f, (long long)N * H);
        (void)hipMemsetAsync(ssum, 0, (size_t)N * H * 4, stream);
        (void)hipMemsetAsync(aggbuf, 0, (size_t)N * H * C * 4, stream);
        gat_scores_kernel<<<cdiv(N * H, 256), 256, 0, stream>>>(hbuf, asv, adv, ssrc, sdst, N, H, C);
        gat_edge_max_kernel<<<cdiv(ET * H, 256), 256, 0, stream>>>(ei, E, ET, H, ssrc, sdst, mx);
        gat_edge_exp_kernel<<<cdiv(ET * H, 256), 256, 0, stream>>>(ei, E, ET, H, ssrc, sdst, mx, alpha, ssum);
        gat_alpha_norm_kernel<<<cdiv(ET * H, 256), 256, 0, stream>>>(ei, E, ET, H, ssum, alpha);
        gat_agg_kernel<<<cdiv((long long)ET * H * C / 4, 256), 256, 0, stream>>>(ei, E, ET, H, C, hbuf, alpha, aggbuf);
        gat_post_kernel<<<cdiv(N * C, 256), 256, 0, stream>>>(aggbuf, bias, outpre, N, H, C);
    };

    // ---- GAT1 (4 heads x 128) -> LN -> relu ----
    gemm4(xn_h, W1t, nullptr, bigA, nullptr, N, 512, 512, 0);        // h1 = xn @ W1
    gat_stage(bigA, as1, ad1, b1, 4, 128, bigB, x1pre);
    ln_kernel<<<cdiv(N, 8), 256, 0, stream>>>(x1pre, nullptr, g1, be1, nullptr, x1_h, N, 128, 1);

    // ---- GCN1 -> relu ----
    gemm4(x1_h, Wg1t, nullptr, bigA, nullptr, N, 128, 128, 0);       // g1h = x1 @ Wg1
    (void)hipMemsetAsync(bigB, 0, (size_t)N * 128 * 4, stream);
    gcn_agg_kernel<<<cdiv((long long)ET * 128 / 4, 256), 256, 0, stream>>>(ei, E, ET, 128, bigA, dis, bigB);
    gcn_post_kernel<<<cdiv(N * 128, 256), 256, 0, stream>>>(bigB, bg1, x1g_h, 128, cat_h, 192, N, 128);

    // ---- GAT2 (2 heads x 64) -> LN -> relu ----
    gemm4(x1g_h, W2t, nullptr, bigA, nullptr, N, 128, 128, 0);       // h2 = x1g @ W2
    gat_stage(bigA, as2, ad2, b2, 2, 64, bigB, x2pre);
    ln_kernel<<<cdiv(N, 8), 256, 0, stream>>>(x2pre, nullptr, g2, be2, nullptr, x2_h, N, 64, 1);

    // ---- GCN2 -> relu (writes into concat cols 128..191) ----
    gemm4(x2_h, Wg2t, nullptr, bigA, nullptr, N, 64, 64, 0);         // g2h = x2 @ Wg2
    (void)hipMemsetAsync(bigB, 0, (size_t)N * 64 * 4, stream);
    gcn_agg_kernel<<<cdiv((long long)ET * 64 / 4, 256), 256, 0, stream>>>(ei, E, ET, 64, bigA, dis, bigB);
    gcn_post_kernel<<<cdiv(N * 64, 256), 256, 0, stream>>>(bigB, bg2, cat_h + 128, 192, nullptr, 0, N, 64);

    // ---- skip + fusion -> LN ----
    gemm4(xn_h, Wskt, bsk, skipb, nullptr, N, 512, 64, 1);           // skip = relu(xn@Ws + bs)
    gemm4(cat_h, Wft, bf, fusedb, nullptr, N, 192, 64, 1);           // fused = relu(cat@Wf + bf)
    ln_kernel<<<cdiv(N, 8), 256, 0, stream>>>(fusedb, skipb, g3, be3, nullptr, fin_h, N, 64, 0);

    // ---- classifier: Linear -> BN(eval) -> relu -> Linear ----
    gemm2(fin_h, Wc1t, bc1, hc1, nullptr, N, 64, 32, 0);
    bn_relu_f32_kernel<<<cdiv((long long)N * 32, 256), 256, 0, stream>>>(
        hc1, gbn, bbn, rmb, rvb, hc1r, (long long)N * 32, 32);
    final_gemm_kernel<<<cdiv(N * 5, 256), 256, 0, stream>>>(hc1r, Wc2, bc2, (float*)d_out, N);
}
